// TransformerOCModel_24172075942494
// MI455X (gfx1250) — compile-verified, matrix operand order checked by reading, not run on hardware
//
#include <hip/hip_runtime.h>
#include <hip/hip_bf16.h>
#include <math.h>

// ---------------------------------------------------------------------------
// Problem constants (match reference)
// ---------------------------------------------------------------------------
#define NN   2048
#define GG   16
#define IND  16
#define DD   256
#define NH   8
#define HD   32
#define FF   1024
#define LL   4
#define LAT  32
#define BH   128

typedef __attribute__((ext_vector_type(16))) _Float16 v16h;
typedef __attribute__((ext_vector_type(8)))  float    v8f;

// ---------------------------------------------------------------------------
// WMMA fragment layout helpers (ISA 16-bit 16x32 A / 32x16 B mapping).
// Within a 32-wide K-slice, local K index kl maps to (half, element e):
//   group = kl/16, w16 = kl%16, half = w16/8, rem = w16%8
//   e = 2*(group*4 + rem/2) + rem%2        (element within the 16-half frag)
// A fragment lane = half*16 + (M%16);  B fragment lane = half*16 + (N%16).
// Tiles of 512 halves (32 lanes x 16 halves), stored lane-major so each lane
// reads one contiguous 32-byte run.
// ---------------------------------------------------------------------------

// fp32 [M,K] row-major -> swizzled f16 A tiles, tile index (tm*nkt + kt)
__global__ __launch_bounds__(256) void swizzleA(const float* __restrict__ src,
                                                _Float16* __restrict__ dst,
                                                int M, int K) {
    int idx = blockIdx.x * 256 + threadIdx.x;
    if (idx >= M * K) return;
    const int m = idx / K, k = idx % K;
    const int tm = m >> 4, l16 = m & 15;
    const int kt = k >> 5, kl = k & 31;
    const int group = kl >> 4, w16 = kl & 15, half = w16 >> 3, rem = w16 & 7;
    const int e = (((group << 2) + (rem >> 1)) << 1) + (rem & 1);
    const int lane = (half << 4) + l16;
    dst[((size_t)(tm * (K >> 5) + kt) << 9) + (lane << 4) + e] = (_Float16)src[idx];
}

// fp32 [K,N] row-major -> swizzled f16 B tiles, tile index (kt*ntn + tn)
__global__ __launch_bounds__(256) void swizzleB(const float* __restrict__ src,
                                                _Float16* __restrict__ dst,
                                                int K, int N) {
    int idx = blockIdx.x * 256 + threadIdx.x;
    if (idx >= K * N) return;
    const int k = idx / N, ncol = idx % N;
    const int kt = k >> 5, kl = k & 31;
    const int tn = ncol >> 4, l16 = ncol & 15;
    const int group = kl >> 4, w16 = kl & 15, half = w16 >> 3, rem = w16 & 7;
    const int e = (((group << 2) + (rem >> 1)) << 1) + (rem & 1);
    const int lane = (half << 4) + l16;
    dst[((size_t)(kt * (N >> 4) + tn) << 9) + (lane << 4) + e] = (_Float16)src[idx];
}

// ---------------------------------------------------------------------------
// WMMA GEMM on pre-swizzled fragments, register double-buffered.
// One wave computes a 16x64 strip of C: 4 accumulators. Pipeline: issue all
// 10 b128 loads for K-tile kt+1, then run the 4 WMMAs of tile kt, so the
// matrix pipe overlaps the memory latency instead of s_wait_loadcnt 0 per op.
// Requires K % 32 == 0 and (Nn/16) % 4 == 0 (true: K in {256,1024},
// Nn in {768,256,1024}).
// ---------------------------------------------------------------------------
__global__ __launch_bounds__(32) void gemm_wmma_sw(const _Float16* __restrict__ A,
                                                   const _Float16* __restrict__ B,
                                                   const float* __restrict__ bias,
                                                   float* __restrict__ C,
                                                   int M, int Nn, int K, int relu) {
    const int nkt = K >> 5;          // K tiles of 32
    const int ntn = Nn >> 4;         // N tiles of 16
    const int ntg = ntn >> 2;        // groups of 4 N-tiles
    const int tm   = blockIdx.x / ntg;
    const int tg   = blockIdx.x % ntg;
    const int lane = threadIdx.x;    // 0..31, wave32
    const int half = lane >> 4;
    const int l16  = lane & 15;

    v8f acc[4];
#pragma unroll
    for (int j = 0; j < 4; ++j) {
        const float bv = bias[(tg * 4 + j) * 16 + l16];
#pragma unroll
        for (int i = 0; i < 8; ++i) acc[j][i] = bv;
    }

    const _Float16* Ap = A + (((size_t)tm * nkt) << 9) + (lane << 4);
    const _Float16* Bp = B + (((size_t)tg * 4) << 9) + (lane << 4);
    const size_t bstep = (size_t)ntn << 9;   // halves between K-tile rows of B

    v16h a[2], bb[2][4];
    a[0] = *(const v16h*)Ap;
#pragma unroll
    for (int j = 0; j < 4; ++j) bb[0][j] = *(const v16h*)(Bp + ((size_t)j << 9));

#pragma unroll 2
    for (int kt = 0; kt < nkt - 1; ++kt) {
        const int cur = kt & 1, nxt = cur ^ 1;
        // prefetch next K-tile fragments (loads in flight during WMMAs below)
        a[nxt] = *(const v16h*)(Ap + ((size_t)(kt + 1) << 9));
        const _Float16* Bn = Bp + (size_t)(kt + 1) * bstep;
#pragma unroll
        for (int j = 0; j < 4; ++j) bb[nxt][j] = *(const v16h*)(Bn + ((size_t)j << 9));
        // consume current K-tile
#pragma unroll
        for (int j = 0; j < 4; ++j) {
            acc[j] = __builtin_amdgcn_wmma_f32_16x16x32_f16(
                false, a[cur], false, bb[cur][j], (short)0, acc[j], false, false);
        }
    }
    {   // epilogue: last K-tile
        const int cur = (nkt - 1) & 1;
#pragma unroll
        for (int j = 0; j < 4; ++j) {
            acc[j] = __builtin_amdgcn_wmma_f32_16x16x32_f16(
                false, a[cur], false, bb[cur][j], (short)0, acc[j], false, false);
        }
    }

#pragma unroll
    for (int j = 0; j < 4; ++j) {
        const int col = (tg * 4 + j) * 16 + l16;
#pragma unroll
        for (int i = 0; i < 8; ++i) {
            float v = acc[j][i];
            if (relu) v = fmaxf(v, 0.0f);
            C[(size_t)(tm * 16 + i + (half << 3)) * Nn + col] = v;   // D: M = i + 8*half
        }
    }
}

// ---------------------------------------------------------------------------
// Input projection: h = x @ Wi + bi   (K = 16 is too small for WMMA; VALU)
// ---------------------------------------------------------------------------
__global__ __launch_bounds__(256) void input_proj(const float* __restrict__ x,
                                                  const float* __restrict__ Wi,
                                                  const float* __restrict__ bi,
                                                  float* __restrict__ h) {
    int idx = blockIdx.x * 256 + threadIdx.x;
    if (idx >= NN * DD) return;
    int n = idx >> 8, d = idx & 255;
    float acc = bi[d];
#pragma unroll
    for (int k = 0; k < IND; ++k) acc = fmaf(x[n * IND + k], Wi[k * DD + d], acc);
    h[idx] = acc;
}

// ---------------------------------------------------------------------------
// Graph ranges: goff[g] = lower_bound(batch, g), goff[G] = N  (batch sorted)
// ---------------------------------------------------------------------------
__global__ void graph_offsets(const int* __restrict__ batch, int* __restrict__ goff) {
    int g = threadIdx.x;
    if (g > GG) return;
    int lo = 0, hi = NN;
    while (lo < hi) {
        int mid = (lo + hi) >> 1;
        if (batch[mid] < g) lo = mid + 1; else hi = mid;
    }
    goff[g] = lo;
}

// ---------------------------------------------------------------------------
// Block-diagonal attention, online softmax. One wave per (node, head);
// lane = head-dim element. qkv layout: [N, 3*D] with q|k|v slices.
// ---------------------------------------------------------------------------
__global__ __launch_bounds__(32) void attention(const float* __restrict__ qkv,
                                                const int* __restrict__ batch,
                                                const int* __restrict__ goff,
                                                float* __restrict__ out) {
    const int wid  = blockIdx.x;       // n*NH + h
    const int n    = wid >> 3;
    const int h    = wid & 7;
    const int lane = threadIdx.x;      // 0..31 == head-dim index
    const int g    = batch[n];
    const int s    = goff[g], e = goff[g + 1];

    const float scale = 0.17677669529663689f;  // 1/sqrt(32)
    const float qd = qkv[n * (3 * DD) + h * HD + lane] * scale;

    float m = -1e30f, ssum = 0.0f, o = 0.0f;
    for (int j = s; j < e; ++j) {
        const float kd = qkv[j * (3 * DD) + DD + h * HD + lane];
        float p = qd * kd;
#pragma unroll
        for (int off = 16; off; off >>= 1) p += __shfl_xor(p, off, 32);
        const float nm   = fmaxf(m, p);
        const float corr = __expf(m - nm);
        const float w    = __expf(p - nm);
        const float vd   = qkv[j * (3 * DD) + 2 * DD + h * HD + lane];
        o    = o * corr + w * vd;
        ssum = ssum * corr + w;
        m    = nm;
    }
    out[n * DD + h * HD + lane] = o / ssum;
}

// ---------------------------------------------------------------------------
// h = LayerNorm(x + r) * s + b   (one 256-thread block per row, D == 256)
// ---------------------------------------------------------------------------
__global__ __launch_bounds__(256) void add_layernorm(const float* __restrict__ r,
                                                     const float* __restrict__ x,
                                                     const float* __restrict__ s,
                                                     const float* __restrict__ b,
                                                     float* __restrict__ hout) {
    __shared__ float red[256];
    const int row = blockIdx.x, t = threadIdx.x;
    const float v = r[row * DD + t] + x[row * DD + t];

    red[t] = v; __syncthreads();
    for (int o = 128; o; o >>= 1) { if (t < o) red[t] += red[t + o]; __syncthreads(); }
    const float mean = red[0] * (1.0f / 256.0f);
    __syncthreads();

    const float d = v - mean;
    red[t] = d * d; __syncthreads();
    for (int o = 128; o; o >>= 1) { if (t < o) red[t] += red[t + o]; __syncthreads(); }
    const float rstd = rsqrtf(red[0] * (1.0f / 256.0f) + 1e-5f);

    hout[row * DD + t] = d * rstd * s[t] + b[t];
}

// ---------------------------------------------------------------------------
// Latent head: lat = normalize(h @ lat_w + lat_b) * sqrt(LAT)
// ---------------------------------------------------------------------------
__global__ __launch_bounds__(32) void lat_head(const float* __restrict__ h,
                                               const float* __restrict__ w,
                                               const float* __restrict__ b,
                                               float* __restrict__ lat) {
    const int n = blockIdx.x, d = threadIdx.x;
    float acc = b[d];
    for (int k = 0; k < DD; ++k) acc = fmaf(h[n * DD + k], w[k * LAT + d], acc);
    float sq = acc * acc;
#pragma unroll
    for (int o = 16; o; o >>= 1) sq += __shfl_xor(sq, o, 32);
    const float nrm = fmaxf(sqrtf(sq), 1e-12f);
    lat[n * LAT + d] = acc / nrm * 5.656854249492381f;  // sqrt(32)
}

// ---------------------------------------------------------------------------
// Beta head: beta = clip(sigmoid(relu(h @ b1_w + b1_b) @ b2_w + b2_b))
// ---------------------------------------------------------------------------
__global__ __launch_bounds__(128) void beta_head(const float* __restrict__ h,
                                                 const float* __restrict__ w1,
                                                 const float* __restrict__ b1,
                                                 const float* __restrict__ w2,
                                                 const float* __restrict__ b2,
                                                 float* __restrict__ beta) {
    __shared__ float red[128];
    const int n = blockIdx.x, t = threadIdx.x;
    float acc = b1[t];
    for (int k = 0; k < DD; ++k) acc = fmaf(h[n * DD + k], w1[k * BH + t], acc);
    acc = fmaxf(acc, 0.0f);
    red[t] = acc * w2[t]; __syncthreads();
    for (int o = 64; o; o >>= 1) { if (t < o) red[t] += red[t + o]; __syncthreads(); }
    if (t == 0) {
        const float z = red[0] + b2[0];
        float sg = 1.0f / (1.0f + __expf(-z));
        sg = fminf(fmaxf(sg, 1e-6f), 1.0f - 1e-6f);
        beta[n] = sg;
    }
}

// ---------------------------------------------------------------------------
// Host orchestration
// ---------------------------------------------------------------------------
static inline size_t alignup(size_t x) { return (x + 255) & ~(size_t)255; }

extern "C" void kernel_launch(void* const* d_in, const int* in_sizes, int n_in,
                              void* d_out, int out_size, void* d_ws, size_t ws_size,
                              hipStream_t stream) {
    (void)in_sizes; (void)n_in; (void)out_size; (void)ws_size;

    const float* x      = (const float*)d_in[0];
    /* coords d_in[1] unused */
    const int*   batch  = (const int*)  d_in[2];
    const float* Wi     = (const float*)d_in[3];
    const float* bi     = (const float*)d_in[4];
    const float* qkv_w  = (const float*)d_in[5];
    const float* qkv_b  = (const float*)d_in[6];
    const float* out_w  = (const float*)d_in[7];
    const float* out_b  = (const float*)d_in[8];
    const float* ff1_w  = (const float*)d_in[9];
    const float* ff1_b  = (const float*)d_in[10];
    const float* ff2_w  = (const float*)d_in[11];
    const float* ff2_b  = (const float*)d_in[12];
    const float* ln1_s  = (const float*)d_in[13];
    const float* ln1_b  = (const float*)d_in[14];
    const float* ln2_s  = (const float*)d_in[15];
    const float* ln2_b  = (const float*)d_in[16];
    const float* lat_w  = (const float*)d_in[17];
    const float* lat_b  = (const float*)d_in[18];
    const float* b1_w   = (const float*)d_in[19];
    const float* b1_b   = (const float*)d_in[20];
    const float* b2_w   = (const float*)d_in[21];
    const float* b2_b   = (const float*)d_in[22];

    // --- workspace carve-out (all 256B aligned) -----------------------------
    char* ws = (char*)d_ws;
    size_t off = 0;
    float*    h_buf   = (float*)   (ws + off); off = alignup(off + (size_t)NN * DD * 4);
    _Float16* h16     = (_Float16*)(ws + off); off = alignup(off + (size_t)NN * DD * 2);
    float*    qkv     = (float*)   (ws + off); off = alignup(off + (size_t)NN * 3 * DD * 4);
    float*    attn_o  = (float*)   (ws + off); off = alignup(off + (size_t)NN * DD * 4);
    _Float16* attn16  = (_Float16*)(ws + off); off = alignup(off + (size_t)NN * DD * 2);
    float*    tmp     = (float*)   (ws + off); off = alignup(off + (size_t)NN * DD * 4);
    float*    ffb     = (float*)   (ws + off); off = alignup(off + (size_t)NN * FF * 4);
    _Float16* ff16    = (_Float16*)(ws + off); off = alignup(off + (size_t)NN * FF * 2);
    _Float16* qkvw16  = (_Float16*)(ws + off); off = alignup(off + (size_t)LL * DD * 3 * DD * 2);
    _Float16* outw16  = (_Float16*)(ws + off); off = alignup(off + (size_t)LL * DD * DD * 2);
    _Float16* ff1w16  = (_Float16*)(ws + off); off = alignup(off + (size_t)LL * DD * FF * 2);
    _Float16* ff2w16  = (_Float16*)(ws + off); off = alignup(off + (size_t)LL * FF * DD * 2);
    int*      goff    = (int*)     (ws + off); off = alignup(off + 32 * 4);

    float* beta_out = (float*)d_out;           // [N]
    float* lat_out  = (float*)d_out + NN;      // [N, LAT]

    const int ND = NN * DD, NFF = NN * FF;

    // weight swizzles into WMMA B-fragment tile layout (per layer slice)
    for (int l = 0; l < LL; ++l) {
        swizzleB<<<(DD * 3 * DD + 255) / 256, 256, 0, stream>>>(
            qkv_w + (size_t)l * DD * 3 * DD, qkvw16 + (size_t)l * DD * 3 * DD, DD, 3 * DD);
        swizzleB<<<(DD * DD + 255) / 256, 256, 0, stream>>>(
            out_w + (size_t)l * DD * DD, outw16 + (size_t)l * DD * DD, DD, DD);
        swizzleB<<<(DD * FF + 255) / 256, 256, 0, stream>>>(
            ff1_w + (size_t)l * DD * FF, ff1w16 + (size_t)l * DD * FF, DD, FF);
        swizzleB<<<(FF * DD + 255) / 256, 256, 0, stream>>>(
            ff2_w + (size_t)l * FF * DD, ff2w16 + (size_t)l * FF * DD, FF, DD);
    }

    graph_offsets<<<1, 32, 0, stream>>>(batch, goff);
    input_proj<<<(ND + 255) / 256, 256, 0, stream>>>(x, Wi, bi, h_buf);

    const int mt = NN / 16;  // 128 M-tiles
    for (int l = 0; l < LL; ++l) {
        // --- self attention block ---
        swizzleA<<<(ND + 255) / 256, 256, 0, stream>>>(h_buf, h16, NN, DD);
        gemm_wmma_sw<<<mt * (3 * DD / 64), 32, 0, stream>>>(
            h16, qkvw16 + (size_t)l * DD * 3 * DD, qkv_b + l * 3 * DD, qkv,
            NN, 3 * DD, DD, 0);
        attention<<<NN * NH, 32, 0, stream>>>(qkv, batch, goff, attn_o);
        swizzleA<<<(ND + 255) / 256, 256, 0, stream>>>(attn_o, attn16, NN, DD);
        gemm_wmma_sw<<<mt * (DD / 64), 32, 0, stream>>>(
            attn16, outw16 + (size_t)l * DD * DD, out_b + l * DD, tmp,
            NN, DD, DD, 0);
        add_layernorm<<<NN, 256, 0, stream>>>(tmp, h_buf, ln1_s + l * DD, ln1_b + l * DD, h_buf);

        // --- feed forward block ---
        swizzleA<<<(ND + 255) / 256, 256, 0, stream>>>(h_buf, h16, NN, DD);
        gemm_wmma_sw<<<mt * (FF / 64), 32, 0, stream>>>(
            h16, ff1w16 + (size_t)l * DD * FF, ff1_b + l * FF, ffb,
            NN, FF, DD, 1 /*relu*/);
        swizzleA<<<(NFF + 255) / 256, 256, 0, stream>>>(ffb, ff16, NN, FF);
        gemm_wmma_sw<<<mt * (DD / 64), 32, 0, stream>>>(
            ff16, ff2w16 + (size_t)l * FF * DD, ff2_b + l * DD, tmp,
            NN, DD, FF, 0);
        add_layernorm<<<NN, 256, 0, stream>>>(tmp, h_buf, ln2_s + l * DD, ln2_b + l * DD, h_buf);
    }

    // --- output heads ---
    lat_head<<<NN, 32, 0, stream>>>(h_buf, lat_w, lat_b, lat_out);
    beta_head<<<NN, 128, 0, stream>>>(h_buf, b1_w, b1_b, b2_w, b2_b, beta_out);
}
